// Decoder_57526791962673
// MI455X (gfx1250) — matrix operand order repaired and checked by reading, not compiled, not run on hardware
//
#include <hip/hip_runtime.h>
#include <hip/hip_bf16.h>
#include <math.h>
#include <stdint.h>

// ---------------------------------------------------------------------------
// Attention-LSTM decoder for MI455X (gfx1250, wave32, WMMA + TDM).
// GEMMs: bf16 operands, f32 accumulate via v_wmma_f32_16x16x32_bf16.
// Output projection stages W_out through LDS with tensor_load_to_lds.
// ---------------------------------------------------------------------------

typedef __bf16 bf16;
typedef __attribute__((ext_vector_type(8)))  bf16     v8bf;
typedef __attribute__((ext_vector_type(16))) bf16     v16bf;
typedef __attribute__((ext_vector_type(8)))  float    v8f;
typedef __attribute__((ext_vector_type(4)))  uint32_t v4u;
typedef __attribute__((ext_vector_type(4)))  int      v4i;
typedef __attribute__((ext_vector_type(8)))  int      v8i;

static constexpr int Bc  = 32;      // batch
static constexpr int Tc  = 800;     // encoder time
static constexpr int Dc  = 512;     // encoder dim (= att_odim)
static constexpr int Ec  = 256;     // embedding dim
static constexpr int Hc  = 512;     // LSTM hidden
static constexpr int Ac  = 512;     // attention hidden
static constexpr int Vc  = 10000;   // vocab
static constexpr int TDc = 100;     // decode steps
static constexpr int KXc = Ec + Dc + Hc;   // 1280 : [emb | ctx | z]
static constexpr int Gc  = 4 * Hc;         // 2048 : gates
static constexpr int Yc  = Hc + Dc;        // 1024 : [z | ctx]

// ------------------------- WMMA fragment helpers ---------------------------
__device__ __forceinline__ v16bf load_a_frag(const bf16* p) {
  v8bf lo = *(const v8bf*)(p);
  v8bf hi = *(const v8bf*)(p + 16);
  return __builtin_shufflevector(lo, hi, 0,1,2,3,4,5,6,7,8,9,10,11,12,13,14,15);
}
__device__ __forceinline__ v8f wmma_bf16(v16bf a, v16bf b, v8f c) {
  return __builtin_amdgcn_wmma_f32_16x16x32_bf16(false, a, false, b,
                                                 (short)0, c, false, false);
}

// ------------------------------ generic GEMM -------------------------------
// C[M,N] = A[M,K](bf16 rm) * B[N,K](bf16 rm)^T ; one 16x16 tile per wave.
template <bool OUT_BF16>
__global__ __launch_bounds__(256) void k_wmma_gemm(
    const bf16* __restrict__ Am, int lda,
    const bf16* __restrict__ Bm, int ldb,
    void* __restrict__ Cp, int ldc,
    int Mtiles, int Ntiles, int K,
    const float* __restrict__ bias)
{
  int wid = blockIdx.x * (blockDim.x >> 5) + (threadIdx.x >> 5);
  if (wid >= Mtiles * Ntiles) return;           // wave-uniform; EXEC stays full
  int m0   = (wid / Ntiles) * 16;
  int n0   = (wid % Ntiles) * 16;
  int lane = threadIdx.x & 31;
  int hl   = lane >> 4;
  int l15  = lane & 15;

  const bf16* ar = Am + (size_t)(m0 + l15) * lda + hl * 8;
  const bf16* br = Bm + (size_t)(n0 + l15) * ldb + hl * 8;

  v8f acc = {};
  for (int k0 = 0; k0 < K; k0 += 32) {
    __builtin_prefetch(br + k0 + 128, 0, 0);
    v16bf a = load_a_frag(ar + k0);
    v16bf b = load_a_frag(br + k0);
    acc = wmma_bf16(a, b, acc);
  }
  int   cm = m0 + hl * 8;
  int   cn = n0 + l15;
  float bv = bias ? bias[cn] : 0.0f;
  if (OUT_BF16) {
    bf16* C = (bf16*)Cp;
#pragma unroll
    for (int i = 0; i < 8; ++i) C[(size_t)(cm + i) * ldc + cn] = (bf16)(acc[i] + bv);
  } else {
    float* C = (float*)Cp;
#pragma unroll
    for (int i = 0; i < 8; ++i) C[(size_t)(cm + i) * ldc + cn] = acc[i] + bv;
  }
}

// --------------------------- M=32 GEMM (2 M-tiles/wave) --------------------
// Each wave computes both 16-row M-tiles for one N-tile: B fragment loaded
// once per K-chunk, two WMMAs -> halves B-side L2 traffic.
__global__ __launch_bounds__(256) void k_wmma_gemm_m32(
    const bf16* __restrict__ Am, int lda,
    const bf16* __restrict__ Bm, int ldb,
    float* __restrict__ C, int ldc,
    int Ntiles, int K,
    const float* __restrict__ bias)
{
  int wid = blockIdx.x * (blockDim.x >> 5) + (threadIdx.x >> 5);
  if (wid >= Ntiles) return;
  int n0   = wid * 16;
  int lane = threadIdx.x & 31;
  int hl   = lane >> 4;
  int l15  = lane & 15;

  const bf16* a0 = Am + (size_t)l15 * lda + hl * 8;          // rows 0..15
  const bf16* a1 = Am + (size_t)(16 + l15) * lda + hl * 8;   // rows 16..31
  const bf16* br = Bm + (size_t)(n0 + l15) * ldb + hl * 8;

  v8f acc0 = {}, acc1 = {};
  for (int k0 = 0; k0 < K; k0 += 32) {
    __builtin_prefetch(br + k0 + 128, 0, 0);
    v16bf b  = load_a_frag(br + k0);
    v16bf a  = load_a_frag(a0 + k0);
    v16bf a2 = load_a_frag(a1 + k0);
    acc0 = wmma_bf16(a,  b, acc0);
    acc1 = wmma_bf16(a2, b, acc1);
  }
  int   cm = hl * 8;
  int   cn = n0 + l15;
  float bv = bias ? bias[cn] : 0.0f;
#pragma unroll
  for (int i = 0; i < 8; ++i) {
    C[(size_t)(cm + i)      * ldc + cn] = acc0[i] + bv;
    C[(size_t)(cm + i + 16) * ldc + cn] = acc1[i] + bv;
  }
}

// ------------------- logits GEMM with TDM LDS staging ----------------------
// Block = 8 waves = 8 N-tiles = 128 rows of W_out. K processed in 128-elem
// blocks; wave 0 DMA-loads the [128 x 128] bf16 slab into LDS with the
// Tensor Data Mover (16B pad every 256B -> 272B row stride, bank-spread),
// s_wait_tensorcnt, then all waves compute from LDS (ds_load_b128).
__global__ __launch_bounds__(256) void k_logits_tdm(
    const bf16* __restrict__ Am,          // Ybf [32][1024]
    const bf16* __restrict__ Wout,        // [10000][1024] bf16
    float* __restrict__ C,                // [32][10000]
    const float* __restrict__ bias)
{
  constexpr int KB   = 128;               // K block (elements)
  constexpr int ROWS = 128;               // W_out rows per block
  constexpr int RPAD = 16;                // LDS pad bytes per row (TDM pad)
  constexpr int ROWB = KB * 2 + RPAD;     // 272B padded LDS row stride
  __shared__ __align__(16) char smem[ROWS * ROWB];

  const int wv   = threadIdx.x >> 5;      // 0..7
  const int lane = threadIdx.x & 31;
  const int hl   = lane >> 4;
  const int l15  = lane & 15;
  const int n0   = blockIdx.x * ROWS + wv * 16;

  const bf16* a0 = Am + (size_t)l15 * Yc + hl * 8;
  const bf16* a1 = Am + (size_t)(16 + l15) * Yc + hl * 8;

  const uint32_t ldsBase  = (uint32_t)(uintptr_t)(void*)smem; // low32 = LDS addr
  const uint32_t rowsLeft = (uint32_t)(Vc - blockIdx.x * ROWS); // >= 16

  v8f acc0 = {}, acc1 = {};
  for (int kb = 0; kb < Yc; kb += KB) {
    if (wv == 0) {
      uint64_t ga = (uint64_t)(uintptr_t)(Wout + (size_t)blockIdx.x * ROWS * Yc + kb);
      // D# group 0: count=1 | lds_addr | global_addr(56:0) | type=2
      v4u g0 = { 1u, ldsBase, (uint32_t)ga,
                 (uint32_t)(ga >> 32) | (2u << 30) };
      // D# group 1: data_size=2B, pad_enable, pad every 64 dwords (code 5),
      // pad amount 4 dwords (code 3); dims/strides per §8.4 bit layout.
      uint32_t td0 = (uint32_t)(Yc - kb);           // tensor_dim0 (cols left)
      uint32_t td1 = rowsLeft;                      // tensor_dim1 (rows left)
      uint32_t s0  = (uint32_t)Yc;                  // tensor_dim0_stride
      uint32_t f   = (1u << 16) | (1u << 20) | (5u << 22) | (3u << 25);
      v8i g1 = { (int)f,
                 (int)((td0 & 0xffffu) << 16),
                 (int)((td0 >> 16) | ((td1 & 0xffffu) << 16)),
                 (int)((td1 >> 16) | ((uint32_t)KB << 16)),   // tile_dim0=128
                 (int)((uint32_t)ROWS),                       // tile_dim1=128
                 (int)s0, 0, 0 };
      v4i gz = { 0, 0, 0, 0 };
#if __clang_major__ >= 23
      v8i gz8 = { 0, 0, 0, 0, 0, 0, 0, 0 };
      __builtin_amdgcn_tensor_load_to_lds(g0, g1, gz, gz, gz8, 0);
#else
      __builtin_amdgcn_tensor_load_to_lds(g0, g1, gz, gz, 0);
#endif
      __builtin_amdgcn_s_wait_tensorcnt(0);
    }
    __syncthreads();
    const char* brow = smem + (size_t)(wv * 16 + l15) * ROWB + hl * 16;
#pragma unroll
    for (int kk = 0; kk < KB; kk += 32) {
      v16bf b  = load_a_frag((const bf16*)(brow + kk * 2));
      v16bf a  = load_a_frag(a0 + kb + kk);
      v16bf a2 = load_a_frag(a1 + kb + kk);
      acc0 = wmma_bf16(a,  b, acc0);
      acc1 = wmma_bf16(a2, b, acc1);
    }
    __syncthreads();
  }
  if (n0 < Vc) {                                   // wave-uniform ragged guard
    int   cm = hl * 8;
    int   cn = n0 + l15;
    float bv = bias[cn];
#pragma unroll
    for (int i = 0; i < 8; ++i) {
      C[(size_t)(cm + i)      * Vc + cn] = acc0[i] + bv;
      C[(size_t)(cm + i + 16) * Vc + cn] = acc1[i] + bv;
    }
  }
}

// --------------------------- precompute kernels ----------------------------
__global__ void k_init_state(float* z, float* cc, float* ctx, int* pred)
{
  int idx = blockIdx.x * blockDim.x + threadIdx.x;
  if (idx < Bc * Hc) { z[idx] = 0.f; cc[idx] = 0.f; }
  if (idx < Bc * Dc) ctx[idx] = 0.f;
  if (idx < Bc)      pred[idx] = 1;   // BOS
}

__global__ void k_pack_w(const float* __restrict__ wih, const float* __restrict__ whh,
                         const float* __restrict__ bih, const float* __restrict__ bhh,
                         bf16* __restrict__ wcomb, float* __restrict__ bcomb)
{
  int idx = blockIdx.x * blockDim.x + threadIdx.x;
  if (idx < Gc) bcomb[idx] = bih[idx] + bhh[idx];
  if (idx >= Gc * KXc) return;
  int n = idx / KXc, c = idx % KXc;
  float v = (c < Ec + Dc) ? wih[(size_t)n * (Ec + Dc) + c]
                          : whh[(size_t)n * Hc + (c - (Ec + Dc))];
  wcomb[idx] = (bf16)v;
}

__global__ void k_transpose_bf(const float* __restrict__ src, bf16* __restrict__ dst,
                               int R, int C)   // dst[c][r] = src[r][c]
{
  int idx = blockIdx.x * blockDim.x + threadIdx.x;
  if (idx >= R * C) return;
  int r = idx / C, c = idx % C;
  dst[(size_t)c * R + r] = (bf16)src[idx];
}

__global__ void k_f2bf(const float* __restrict__ src, bf16* __restrict__ dst, int n)
{
  int idx = blockIdx.x * blockDim.x + threadIdx.x;
  if (idx < n) dst[idx] = (bf16)src[idx];
}

// ---------------------------- per-step kernels -----------------------------
__global__ void k_build_xz(const float* __restrict__ emb, const int* __restrict__ pred,
                           const float* __restrict__ ctx, const float* __restrict__ z,
                           bf16* __restrict__ xz)
{
  int idx = blockIdx.x * blockDim.x + threadIdx.x;
  if (idx >= Bc * KXc) return;
  int b = idx / KXc, c = idx % KXc;
  float v;
  if (c < Ec)           v = emb[(size_t)pred[b] * Ec + c];
  else if (c < Ec + Dc) v = ctx[(size_t)b * Dc + (c - Ec)];
  else                  v = z[(size_t)b * Hc + (c - Ec - Dc)];
  xz[idx] = (bf16)v;
}

__global__ void k_lstm_act(const float* __restrict__ gates, float* __restrict__ z,
                           float* __restrict__ cc, bf16* __restrict__ ybf)
{
  int idx = blockIdx.x * blockDim.x + threadIdx.x;
  if (idx >= Bc * Hc) return;
  int b = idx / Hc, h = idx % Hc;
  const float* g = gates + (size_t)b * Gc;
  float ig = 1.f / (1.f + expf(-g[h]));
  float fg = 1.f / (1.f + expf(-g[Hc + h]));
  float gg = tanhf(g[2 * Hc + h]);
  float og = 1.f / (1.f + expf(-g[3 * Hc + h]));
  float cn = fg * cc[idx] + ig * gg;
  float zn = og * tanhf(cn);
  cc[idx] = cn;
  z[idx]  = zn;
  ybf[(size_t)b * Yc + h] = (bf16)zn;   // first half of [z|ctx]
}

__global__ void k_attn_score(const bf16* __restrict__ encproj, const float* __restrict__ q,
                             const float* __restrict__ vatt, float* __restrict__ e)
{
  int idx = blockIdx.x * blockDim.x + threadIdx.x;  // b*Tc + t
  if (idx >= Bc * Tc) return;
  int b = idx / Tc;
  const bf16*  row = encproj + (size_t)idx * Ac;
  const float* qb  = q + (size_t)b * Ac;
  float acc = 0.f;
#pragma unroll 4
  for (int d = 0; d < Ac; ++d)
    acc += vatt[d] * tanhf((float)row[d] + qb[d]);
  e[idx] = acc;
}

__global__ void k_softmax(const float* __restrict__ e, const int* __restrict__ enc_len,
                          float* __restrict__ outws, int step)
{
  int b = blockIdx.x, tid = threadIdx.x;
  __shared__ float sr[256];
  int len = enc_len[b];
  const float* er = e + (size_t)b * Tc;
  float mx = -1e30f;
  for (int t = tid; t < Tc; t += 256) if (t < len) mx = fmaxf(mx, er[t]);
  sr[tid] = mx; __syncthreads();
  for (int s = 128; s > 0; s >>= 1) { if (tid < s) sr[tid] = fmaxf(sr[tid], sr[tid + s]); __syncthreads(); }
  mx = sr[0]; __syncthreads();
  float sum = 0.f;
  for (int t = tid; t < Tc; t += 256) if (t < len) sum += expf(er[t] - mx);
  sr[tid] = sum; __syncthreads();
  for (int s = 128; s > 0; s >>= 1) { if (tid < s) sr[tid] += sr[tid + s]; __syncthreads(); }
  float inv = 1.f / sr[0];
  float* wrow = outws + ((size_t)b * TDc + step) * Tc;
  for (int t = tid; t < Tc; t += 256)
    wrow[t] = (t < len) ? expf(er[t] - mx) * inv : 0.f;
}

__global__ void k_context(const float* __restrict__ outws, const bf16* __restrict__ encbf,
                          float* __restrict__ ctx, bf16* __restrict__ ybf, int step)
{
  int b = blockIdx.x, d = threadIdx.x;          // blockDim = 512
  const float* wrow = outws + ((size_t)b * TDc + step) * Tc;
  const bf16*  eb   = encbf + (size_t)b * Tc * Dc + d;
  float acc = 0.f;
  for (int t = 0; t < Tc; ++t) acc += wrow[t] * (float)eb[(size_t)t * Dc];
  ctx[(size_t)b * Dc + d] = acc;
  ybf[(size_t)b * Yc + Hc + d] = (bf16)acc;     // second half of [z|ctx]
}

__global__ void k_finalize(const float* __restrict__ logits, float* __restrict__ out,
                           int* __restrict__ pred_state, int step)
{
  int b = blockIdx.x, tid = threadIdx.x;
  __shared__ float sv[256];
  __shared__ int   si[256];
  const float* row = logits + (size_t)b * Vc;
  float bestv = -INFINITY; int besti = 0;
  for (int v = tid; v < Vc; v += 256) {
    float x = row[v];
    if (x > bestv) { bestv = x; besti = v; }
  }
  sv[tid] = bestv; si[tid] = besti; __syncthreads();
  for (int s = 128; s > 0; s >>= 1) {
    if (tid < s) {
      float ov = sv[tid + s]; int oi = si[tid + s];
      if (ov > sv[tid] || (ov == sv[tid] && oi < si[tid])) { sv[tid] = ov; si[tid] = oi; }
    }
    __syncthreads();
  }
  float mx = sv[0]; int arg = si[0]; __syncthreads();
  float acc = 0.f;
  for (int v = tid; v < Vc; v += 256) acc += expf(row[v] - mx);
  sv[tid] = acc; __syncthreads();
  for (int s = 128; s > 0; s >>= 1) { if (tid < s) sv[tid] += sv[tid + s]; __syncthreads(); }
  if (tid == 0) {
    out[(size_t)b * TDc + step]            = -logf(sv[0]);      // ys_log_probs
    out[Bc * TDc + (size_t)b * TDc + step] = (float)arg;        // preds (as f32)
    pred_state[b] = arg;
  }
}

// ------------------------------- launcher ----------------------------------
extern "C" void kernel_launch(void* const* d_in, const int* in_sizes, int n_in,
                              void* d_out, int out_size, void* d_ws, size_t ws_size,
                              hipStream_t stream)
{
  (void)in_sizes; (void)n_in; (void)out_size; (void)ws_size;
  const float* enc_pad   = (const float*)d_in[0];
  const float* embedding = (const float*)d_in[1];
  const float* W_ih      = (const float*)d_in[2];
  const float* W_hh      = (const float*)d_in[3];
  const float* b_ih      = (const float*)d_in[4];
  const float* b_hh      = (const float*)d_in[5];
  const float* W_enc     = (const float*)d_in[6];
  const float* W_dec     = (const float*)d_in[7];
  const float* v_att     = (const float*)d_in[8];
  const float* W_out     = (const float*)d_in[9];
  const float* b_out     = (const float*)d_in[10];
  const int*   enc_len   = (const int*)d_in[11];

  char*  ws  = (char*)d_ws;
  size_t off = 0;
  auto alloc = [&](size_t bytes) -> void* {
    void* p = ws + off; off += (bytes + 255) & ~(size_t)255; return p;
  };
  bf16*  Wcomb   = (bf16*)alloc((size_t)Gc * KXc * 2);
  bf16*  WencT   = (bf16*)alloc((size_t)Ac * Dc * 2);
  bf16*  WdecT   = (bf16*)alloc((size_t)Ac * Hc * 2);
  bf16*  WoutBf  = (bf16*)alloc((size_t)Vc * Yc * 2);
  bf16*  EncBf   = (bf16*)alloc((size_t)Bc * Tc * Dc * 2);
  bf16*  EncProj = (bf16*)alloc((size_t)Bc * Tc * Ac * 2);
  float* Bcomb   = (float*)alloc((size_t)Gc * 4);
  float* Z       = (float*)alloc((size_t)Bc * Hc * 4);
  float* Cc      = (float*)alloc((size_t)Bc * Hc * 4);
  float* Ctx     = (float*)alloc((size_t)Bc * Dc * 4);
  int*   Pred    = (int*)  alloc((size_t)Bc * 4);
  bf16*  Xz      = (bf16*)alloc((size_t)Bc * KXc * 2);
  bf16*  Ybf     = (bf16*)alloc((size_t)Bc * Yc * 2);
  float* Gates   = (float*)alloc((size_t)Bc * Gc * 4);
  float* Q       = (float*)alloc((size_t)Bc * Ac * 4);
  float* Escore  = (float*)alloc((size_t)Bc * Tc * 4);
  float* Logits  = (float*)alloc((size_t)Bc * Vc * 4);

  float* out   = (float*)d_out;
  float* outws = out + 2 * Bc * TDc;   // attention-weight slab [B,TD,T]

  auto blocks = [](long n, int bs) { return (int)((n + bs - 1) / bs); };

  // ---- one-time precompute ----
  k_init_state<<<blocks(Bc * Hc, 256), 256, 0, stream>>>(Z, Cc, Ctx, Pred);
  k_pack_w<<<blocks((long)Gc * KXc, 256), 256, 0, stream>>>(W_ih, W_hh, b_ih, b_hh, Wcomb, Bcomb);
  k_transpose_bf<<<blocks((long)Dc * Ac, 256), 256, 0, stream>>>(W_enc, WencT, Dc, Ac);
  k_transpose_bf<<<blocks((long)Hc * Ac, 256), 256, 0, stream>>>(W_dec, WdecT, Hc, Ac);
  k_f2bf<<<blocks((long)Vc * Yc, 256), 256, 0, stream>>>(W_out, WoutBf, Vc * Yc);
  k_f2bf<<<blocks((long)Bc * Tc * Dc, 256), 256, 0, stream>>>(enc_pad, EncBf, Bc * Tc * Dc);

  // enc_proj = enc_pad @ W_enc : M=25600, N=512, K=512, bf16 out
  {
    int Mt = (Bc * Tc) / 16, Nt = Ac / 16;
    k_wmma_gemm<true><<<blocks((long)Mt * Nt, 8), 256, 0, stream>>>(
        EncBf, Dc, WencT, Dc, EncProj, Ac, Mt, Nt, Dc, nullptr);
  }

  // ---- sequential decode ----
  const int logitsBlocks = (Vc + 127) / 128;   // 79 (ragged last handled by TDM OOB)
  for (int step = 0; step < TDc; ++step) {
    k_build_xz<<<blocks((long)Bc * KXc, 256), 256, 0, stream>>>(embedding, Pred, Ctx, Z, Xz);

    // gates = xz @ [W_ih|W_hh]^T + (b_ih+b_hh) : N=2048, K=1280
    k_wmma_gemm_m32<<<blocks(Gc / 16, 8), 256, 0, stream>>>(
        Xz, KXc, Wcomb, KXc, Gates, Gc, Gc / 16, KXc, Bcomb);

    k_lstm_act<<<blocks((long)Bc * Hc, 256), 256, 0, stream>>>(Gates, Z, Cc, Ybf);

    // q = z @ W_dec : N=512, K=512 (A = z-half of Ybf, lda=Yc)
    k_wmma_gemm_m32<<<blocks(Ac / 16, 8), 256, 0, stream>>>(
        Ybf, Yc, WdecT, Hc, Q, Ac, Ac / 16, Hc, nullptr);

    k_attn_score<<<blocks((long)Bc * Tc, 256), 256, 0, stream>>>(EncProj, Q, v_att, Escore);
    k_softmax<<<Bc, 256, 0, stream>>>(Escore, enc_len, outws, step);
    k_context<<<Bc, Dc, 0, stream>>>(outws, EncBf, Ctx, Ybf, step);

    // logits = [z|ctx] @ W_out^T + b_out : TDM-staged through LDS
    k_logits_tdm<<<logitsBlocks, 256, 0, stream>>>(Ybf, WoutBf, Logits, b_out);

    k_finalize<<<Bc, 256, 0, stream>>>(Logits, out, Pred, step);
  }
}
